// GraphModule_30820685316777
// MI455X (gfx1250) — matrix operand order repaired and checked by reading, not compiled, not run on hardware
//
#include <hip/hip_runtime.h>
#include <hip/hip_bf16.h>
#include <math.h>

// ---------------------------------------------------------------------------
// NeZha encoder, 12 layers, on gfx1250 (MI455X).
// Every contraction runs on v_wmma_f32_16x16x32_bf16. Weights/dist_emb are
// bf16-converted once per launch; activations are produced in bf16 at their
// definition site. Large BT GEMMs stage tiles with the Tensor Data Mover
// (double-buffered TENSOR_LOAD_TO_LDS; D# pad fields produce the padded LDS
// row stride). Other GEMMs use uint4 manual staging.
// Rel-pos trick: for S=512 <= MAXPOS the clip is a no-op, so
// pe[l,r,:] = dist[l-r+511,:] is a constant-stride view => both bias
// einsums are batched GEMMs (ldb = -64 / +64) accumulating into scores.
// ---------------------------------------------------------------------------

#define NB 16
#define NS 512
#define NHID 768
#define NHEADS 12
#define NHD 64
#define NFF 3072
#define NL 12
#define NMAXPOS 512

typedef __attribute__((ext_vector_type(16))) __bf16 v16bf;
typedef __attribute__((ext_vector_type(8)))  float  v8f;
typedef __attribute__((ext_vector_type(4)))  unsigned int v4u;
typedef __attribute__((ext_vector_type(8)))  int v8i;
typedef __attribute__((ext_vector_type(4)))  int v4i;

union FragAB { v16bf v; unsigned short u[16]; };
union Pack8  { uint4 v; unsigned short s[8]; };

__device__ __forceinline__ unsigned short f2bf(float f) {
    unsigned int u = __float_as_uint(f);
    u += 0x7FFFu + ((u >> 16) & 1u);   // round-to-nearest-even
    return (unsigned short)(u >> 16);
}

#define BK 64
#define LDSA 72   // padded row stride in shorts (144B = 32 data DW + 4 pad DW)
#define LDSB 72

// TDM: DMA a 2D bf16 tile (rows x 64) from global to LDS, inserting 4 pad
// DWORDs after every 32 data DWORDs (row stride 72 shorts). D# per ISA 8.3/8.4.
__device__ __forceinline__ void tdm_load_2d(unsigned ldsAddr,
                                            const unsigned short* g,
                                            int rows, long strideElems) {
    unsigned long long ga = (unsigned long long)g;
    unsigned long long sb = (unsigned long long)strideElems;  // dim0 stride (elems)
    const unsigned td0 = 1u << 30, td1 = 1u << 20;            // huge: no OOB clip
    v4u g0;
    g0[0] = 1u;                                               // count=1, user mode
    g0[1] = ldsAddr;                                          // lds_addr
    g0[2] = (unsigned)ga;                                     // global_addr[31:0]
    g0[3] = (unsigned)((ga >> 32) & 0x01FFFFFFu) | (2u << 30);// ga[56:32], type=2
    v8i g1;
    g1[0] = (int)((1u << 16) | (1u << 20) | (4u << 22) | (3u << 25));
    //             2B elems     pad_enable   32DW interval  4DW pad
    g1[1] = (int)((td0 & 0xFFFFu) << 16);                     // tensor_dim0 lo
    g1[2] = (int)(((td0 >> 16) & 0xFFFFu) | ((td1 & 0xFFFFu) << 16));
    g1[3] = (int)(((td1 >> 16) & 0xFFFFu) | ((unsigned)BK << 16)); // tile_dim0=64
    g1[4] = (int)((unsigned)rows & 0xFFFFu);                  // tile_dim1, dim2=0
    g1[5] = (int)(unsigned)(sb & 0xFFFFFFFFull);              // dim0_stride lo
    g1[6] = (int)(unsigned)((sb >> 32) & 0xFFFFull);          // dim0_stride hi
    g1[7] = 0;
    v4i z4 = {0, 0, 0, 0};
    v8i z8 = {0, 0, 0, 0, 0, 0, 0, 0};
    __builtin_amdgcn_tensor_load_to_lds(g0, g1, z4, z4, z8, 0);
}

// A fragment: 16x32 bf16 per ISA layout. lane = M(0..15) + 16*half,
// VGPR v holds K = (v<4 ? 2v : 2v+8) + 8*half and K+1 (plus kstep offset).
__device__ __forceinline__ void loadFragA(FragAB& f, const unsigned short* As,
                                          int mbase, int koff, int r16, int half) {
    const unsigned short* row = As + (mbase + r16) * LDSA + koff;
#pragma unroll
    for (int vv = 0; vv < 8; ++vv) {
        int kb = ((vv < 4) ? (2 * vv) : (2 * vv + 8)) + 8 * half;
        f.u[2 * vv]     = row[kb];
        f.u[2 * vv + 1] = row[kb + 1];
    }
}

// B fragment: 32x16 bf16 (col N = lane&15), same K order as A:
// VGPR v holds K = (v<4 ? 2v : 2v+8) + 8*half and K+1.
__device__ __forceinline__ void loadFragB(FragAB& f, const unsigned short* Bs,
                                          int nbase, int koff, int r16, int half) {
    const unsigned short* row = Bs + (nbase + r16) * LDSB + koff;
#pragma unroll
    for (int vv = 0; vv < 8; ++vv) {
        int kb = ((vv < 4) ? (2 * vv) : (2 * vv + 8)) + 8 * half;
        f.u[2 * vv]     = row[kb];
        f.u[2 * vv + 1] = row[kb + 1];
    }
}

// C/D layout: VGPR r -> M = r + 8*half, N = lane&15.
// CBHSD: write [B,NH,S,HD] from m=(b,s), n=(h,d) (compile-time S=512, HD=64).
template<bool GELU_, bool RESID, bool OUTBF, bool CBHSD>
__device__ __forceinline__ void storeTile(v8f acc, void* Cv, const float* bias,
                                          const float* resid, long cOff,
                                          int m0, int n0, int mt, int ntile,
                                          int ldc, int cNst, int r16, int half) {
    int n = n0 + ntile + r16;
    float bv = bias ? bias[n] : 0.0f;
#pragma unroll
    for (int rr = 0; rr < 8; ++rr) {
        int m = m0 + mt + rr + 8 * half;
        long off;
        if (CBHSD)
            off = (long)(m >> 9) * ((long)NHEADS * NS * NHD) + (long)(m & 511) * NHD
                + (long)(n >> 6) * ((long)NS * NHD) + (n & 63);
        else
            off = cOff + (long)m * ldc + (long)n * cNst;
        float val = acc[rr] + bv;
        if (GELU_) val = 0.5f * val * (1.0f + erff(val * 0.70710678118f));
        if (RESID) val += resid[off];
        if (OUTBF) ((unsigned short*)Cv)[off] = f2bf(val);
        else       ((float*)Cv)[off] = val;
    }
}

// C[M,N] = op(A[M,K] x B) + bias (+GELU) (+resid/accum), A/B bf16, acc f32.
//   BT=true : B is [N,K] row-major (ldb may be negative when !TDM)
//   BT=false: B is [K,N] row-major (probs @ V style)
//   TDM=true: double-buffered TENSOR_LOAD_TO_LDS staging (wave 0 issues DMA)
// Per-batch base offsets: (z/div)*outer + (z%div)*inner for A, B, C.
// TBM x TBN block tile, 256 threads = 8 wave32 each owning 32x32.
template<int TBM, int TBN, bool BT, bool GELU_, bool RESID, bool OUTBF,
         bool CBHSD, bool TDM>
__global__ __launch_bounds__(256)
void gemm_wmma(const unsigned short* __restrict__ A,
               const unsigned short* __restrict__ Bm,
               void* Cv, const float* __restrict__ bias, const float* resid,
               int Ksz, int lda, int ldb, int ldc, int cNst,
               int aDiv, long aOuter, long aInner,
               int bDiv, long bOuter, long bInner,
               int cDiv, long cOuter, long cInner) {
    static_assert(TBM * TBN == 8192, "8 waves of 32x32");
    static_assert(!TDM || BT, "TDM staging only for the [N,K] B layout");
    constexpr int NBUF = TDM ? 2 : 1;
    __shared__ unsigned short As[NBUF * TBM * LDSA];
    __shared__ unsigned short Bs[NBUF * TBN * LDSB];
    constexpr int WN  = TBN / 32;   // waves along N
    constexpr int NA  = TBM / 32;   // A uint4 staging loads per thread
    constexpr int NBS = TBN / 32;   // B uint4 staging loads per thread

    int z = blockIdx.z;
    const unsigned short* Ab = A + (long)(z / aDiv) * aOuter + (long)(z % aDiv) * aInner;
    const unsigned short* Bb = Bm + (long)(z / bDiv) * bOuter + (long)(z % bDiv) * bInner;
    long cOff = (long)(z / cDiv) * cOuter + (long)(z % cDiv) * cInner;

    int m0 = blockIdx.y * TBM;
    int n0 = blockIdx.x * TBN;
    int tid = threadIdx.x;
    int lane = tid & 31, w = tid >> 5;
    int half = lane >> 4, r16 = lane & 15;
    int wm = w / WN, wn = w % WN;
    int am0 = wm * 32, bn0 = wn * 32;

    v8f acc00 = {}, acc01 = {}, acc10 = {}, acc11 = {};

    int nk = Ksz / BK;
    if (TDM && tid < 32) {          // prologue: DMA stage 0 into buffer 0
        tdm_load_2d((unsigned)(unsigned long long)&As[0],
                    Ab + (long)m0 * lda, TBM, lda);
        tdm_load_2d((unsigned)(unsigned long long)&Bs[0],
                    Bb + (long)n0 * ldb, TBN, ldb);
    }
    for (int kt = 0; kt < nk; ++kt) {
        int k0 = kt * BK;
        const unsigned short* Acur = As;
        const unsigned short* Bcur = Bs;

        if (TDM) {
            int buf = kt & 1;
            Acur = As + buf * (TBM * LDSA);
            Bcur = Bs + buf * (TBN * LDSB);
            if (tid < 32) {
                if (kt + 1 < nk) {  // DMA next stage into the other buffer
                    int nb = buf ^ 1;
                    tdm_load_2d((unsigned)(unsigned long long)&As[nb * (TBM * LDSA)],
                                Ab + (long)m0 * lda + k0 + BK, TBM, lda);
                    tdm_load_2d((unsigned)(unsigned long long)&Bs[nb * (TBN * LDSB)],
                                Bb + (long)n0 * ldb + k0 + BK, TBN, ldb);
                    // pairs complete in order: <=2 outstanding => stage kt done
                    __builtin_amdgcn_s_wait_tensorcnt(2);
                } else {
                    __builtin_amdgcn_s_wait_tensorcnt(0);
                }
            }
        } else {
            // ---- issue ALL staging loads first (pipeline under one wait) ----
            uint4 fa[NA];                        // A: TBM rows x 8 uint4 cols
#pragma unroll
            for (int i = 0; i < NA; ++i) {
                int idx = tid + i * 256;
                int row = idx >> 3, c8 = (idx & 7) << 3;
                fa[i] = *(const uint4*)(Ab + (long)(m0 + row) * lda + k0 + c8);
            }
            uint4 fb[NBS];                       // B: TBN rows x 8 uint4 cols
            if (BT) {
#pragma unroll
                for (int i = 0; i < NBS; ++i) {
                    int idx = tid + i * 256;
                    int row = idx >> 3, c8 = (idx & 7) << 3;
                    fb[i] = *(const uint4*)(Bb + (long)(n0 + row) * ldb + k0 + c8);
                }
            } else {
#pragma unroll
                for (int i = 0; i < NBS; ++i) {  // (k:BK) x (n8:TBN/8)
                    int idx = tid + i * 256;
                    int kk = idx / (TBN / 8), n8 = (idx % (TBN / 8)) << 3;
                    fb[i] = *(const uint4*)(Bb + (long)(k0 + kk) * ldb + n0 + n8);
                }
            }
            // prefetch next K-tile of A (global_prefetch_b8 on gfx1250)
            if (kt + 1 < nk)
                __builtin_prefetch(Ab + (long)(m0 + (tid >> 1)) * lda + k0 + BK, 0, 1);

            // ---- stage into LDS (pure copies, no conversion) ----
#pragma unroll
            for (int i = 0; i < NA; ++i) {
                int idx = tid + i * 256;
                int row = idx >> 3, c8 = (idx & 7) << 3;
                *(uint4*)&As[row * LDSA + c8] = fa[i];
            }
            if (BT) {
#pragma unroll
                for (int i = 0; i < NBS; ++i) {
                    int idx = tid + i * 256;
                    int row = idx >> 3, c8 = (idx & 7) << 3;
                    *(uint4*)&Bs[row * LDSB + c8] = fb[i];
                }
            } else {
#pragma unroll
                for (int i = 0; i < NBS; ++i) {  // transpose scatter [k][n]->[n][k]
                    int idx = tid + i * 256;
                    int kk = idx / (TBN / 8), n8 = (idx % (TBN / 8)) << 3;
                    Pack8 p; p.v = fb[i];
#pragma unroll
                    for (int j = 0; j < 8; ++j)
                        Bs[(n8 + j) * LDSB + kk] = p.s[j];
                }
            }
        }
        __syncthreads();

        // ---- 2 WMMA K-steps per stage ----
#pragma unroll
        for (int ks = 0; ks < 2; ++ks) {
            int koff = ks * 32;
            FragAB a0, a1, b0, b1;
            loadFragA(a0, Acur, am0,      koff, r16, half);
            loadFragA(a1, Acur, am0 + 16, koff, r16, half);
            loadFragB(b0, Bcur, bn0,      koff, r16, half);
            loadFragB(b1, Bcur, bn0 + 16, koff, r16, half);

            acc00 = __builtin_amdgcn_wmma_f32_16x16x32_bf16(false, a0.v, false, b0.v,
                                                            (short)0, acc00, false, false);
            acc01 = __builtin_amdgcn_wmma_f32_16x16x32_bf16(false, a0.v, false, b1.v,
                                                            (short)0, acc01, false, false);
            acc10 = __builtin_amdgcn_wmma_f32_16x16x32_bf16(false, a1.v, false, b0.v,
                                                            (short)0, acc10, false, false);
            acc11 = __builtin_amdgcn_wmma_f32_16x16x32_bf16(false, a1.v, false, b1.v,
                                                            (short)0, acc11, false, false);
        }
        __syncthreads();
    }

    storeTile<GELU_, RESID, OUTBF, CBHSD>(acc00, Cv, bias, resid, cOff, m0, n0, am0,      bn0,      ldc, cNst, r16, half);
    storeTile<GELU_, RESID, OUTBF, CBHSD>(acc01, Cv, bias, resid, cOff, m0, n0, am0,      bn0 + 16, ldc, cNst, r16, half);
    storeTile<GELU_, RESID, OUTBF, CBHSD>(acc10, Cv, bias, resid, cOff, m0, n0, am0 + 16, bn0,      ldc, cNst, r16, half);
    storeTile<GELU_, RESID, OUTBF, CBHSD>(acc11, Cv, bias, resid, cOff, m0, n0, am0 + 16, bn0 + 16, ldc, cNst, r16, half);
}

// ---------------- fp32 -> bf16 bulk conversion (once per launch) ----------
__global__ __launch_bounds__(256)
void cvt_bf16_kernel(const float* __restrict__ in, unsigned short* __restrict__ out,
                     long n) {
    long i = ((long)blockIdx.x * 256 + threadIdx.x) * 8;
    if (i + 8 > n) return;
    float4 a = *(const float4*)(in + i);
    float4 b = *(const float4*)(in + i + 4);
    Pack8 o;
    o.s[0] = f2bf(a.x); o.s[1] = f2bf(a.y); o.s[2] = f2bf(a.z); o.s[3] = f2bf(a.w);
    o.s[4] = f2bf(b.x); o.s[5] = f2bf(b.y); o.s[6] = f2bf(b.z); o.s[7] = f2bf(b.w);
    *(uint4*)(out + i) = o.v;
}

// ---------------- embedding: h = word_emb[id] + tok_emb[tt] ----------------
__global__ __launch_bounds__(256)
void embed_kernel(const int* __restrict__ ids, const long long* __restrict__ tt,
                  const float* __restrict__ we, const float* __restrict__ te,
                  float* __restrict__ out) {
    long row = blockIdx.x;
    int id = ids[row];
    long long t = tt[row];
    const float* wrow = we + (long)id * NHID;
    const float* trow = te + (long)t * NHID;
    for (int c = threadIdx.x; c < NHID; c += 256)
        out[row * NHID + c] = wrow[c] + trow[c];
}

// ------- LayerNorm over H=768, one block per row; writes bf16 directly -------
__global__ __launch_bounds__(256)
void layernorm_kernel(const float* __restrict__ in, const float* __restrict__ w,
                      const float* __restrict__ b, unsigned short* __restrict__ out) {
    __shared__ float s1[256], s2[256];
    long row = blockIdx.x;
    const float* p = in + row * NHID;
    int tid = threadIdx.x;
    float sum = 0.f, sq = 0.f;
    for (int c = tid; c < NHID; c += 256) { float x = p[c]; sum += x; sq += x * x; }
    s1[tid] = sum; s2[tid] = sq;
    __syncthreads();
    for (int st = 128; st > 0; st >>= 1) {
        if (tid < st) { s1[tid] += s1[tid + st]; s2[tid] += s2[tid + st]; }
        __syncthreads();
    }
    float mu = s1[0] * (1.0f / NHID);
    float var = s2[0] * (1.0f / NHID) - mu * mu;
    float inv = rsqrtf(var + 1e-12f);
    for (int c = tid; c < NHID; c += 256)
        out[row * NHID + c] = f2bf((p[c] - mu) * inv * w[c] + b[c]);
}

// --- softmax(scale*s + mask): fp32 in, bf16 probs out; one wave32 per row ---
__global__ __launch_bounds__(256)
void softmax_kernel(const float* __restrict__ sc, const int* __restrict__ mask,
                    unsigned short* __restrict__ probs) {
    long row = (long)blockIdx.x * 8 + (threadIdx.x >> 5);
    int lane = threadIdx.x & 31;
    int b = (int)(row / ((long)NHEADS * NS));
    const float* p = sc + row * NS;
    unsigned short* po = probs + row * NS;
    const int* mrow = mask + (long)b * NS;
    const float scale = 0.125f;           // 1/sqrt(64)
    float vals[16];
    float mx = -1e30f;
#pragma unroll
    for (int i = 0; i < 16; ++i) {
        int c = lane + i * 32;
        float x = p[c] * scale + (float)(1 - mrow[c]) * (-10000.0f);
        vals[i] = x;
        mx = fmaxf(mx, x);
    }
    for (int o = 16; o > 0; o >>= 1) mx = fmaxf(mx, __shfl_xor(mx, o, 32));
    float s = 0.f;
#pragma unroll
    for (int i = 0; i < 16; ++i) { vals[i] = __expf(vals[i] - mx); s += vals[i]; }
    for (int o = 16; o > 0; o >>= 1) s += __shfl_xor(s, o, 32);
    float inv = 1.0f / s;
#pragma unroll
    for (int i = 0; i < 16; ++i) po[lane + i * 32] = f2bf(vals[i] * inv);
}

// ---------------------------------------------------------------------------
extern "C" void kernel_launch(void* const* d_in, const int* in_sizes, int n_in,
                              void* d_out, int out_size, void* d_ws, size_t ws_size,
                              hipStream_t stream) {
    const int*       input_ids = (const int*)d_in[0];
    const int*       attn_mask = (const int*)d_in[1];
    const long long* tok_type  = (const long long*)d_in[2];
    const float* word_emb   = (const float*)d_in[3];
    const float* tok_emb    = (const float*)d_in[4];
    const float* attn_ln_w  = (const float*)d_in[5];
    const float* attn_ln_b  = (const float*)d_in[6];
    const float* qw = (const float*)d_in[7];
    const float* qb = (const float*)d_in[8];
    const float* kw = (const float*)d_in[9];
    const float* kb = (const float*)d_in[10];
    const float* vw = (const float*)d_in[11];
    const float* vb = (const float*)d_in[12];
    const float* dist_emb   = (const float*)d_in[13];
    const float* attn_out_w = (const float*)d_in[14];
    const float* attn_out_b = (const float*)d_in[15];
    const float* ffn_ln_w   = (const float*)d_in[16];
    const float* ffn_ln_b   = (const float*)d_in[17];
    const float* int_w = (const float*)d_in[18];
    const float* int_b = (const float*)d_in[19];
    const float* out_w = (const float*)d_in[20];
    const float* out_b = (const float*)d_in[21];

    const int M = NB * NS;                 // 8192 rows
    float* h = (float*)d_out;              // residual stream lives in d_out

    const long W1 = (long)NL * NHID * NHID;        //  7,077,888
    const long W2 = (long)NL * NFF * NHID;         // 28,311,552
    const long WD = (long)NL * (2 * NMAXPOS - 1) * NHD;  // 785,664
    unsigned short* bf = (unsigned short*)d_ws;
    unsigned short* qw_bf = bf;                 bf += W1;
    unsigned short* kw_bf = bf;                 bf += W1;
    unsigned short* vw_bf = bf;                 bf += W1;
    unsigned short* ow_bf = bf;                 bf += W1;
    unsigned short* iw_bf = bf;                 bf += W2;
    unsigned short* pw_bf = bf;                 bf += W2;
    unsigned short* dist_bf = bf;               bf += WD + 8;
    unsigned short* x_bf   = bf;                bf += (long)M * NHID;
    unsigned short* q_bf   = bf;                bf += (long)M * NHID;  // [B,NH,S,HD]
    unsigned short* k_bf   = bf;                bf += (long)M * NHID;  // [B,NH,S,HD]
    unsigned short* v_bf   = bf;                bf += (long)M * NHID;  // [B,S,H]
    unsigned short* ctx_bf = bf;                bf += (long)M * NHID;  // [B,S,H]
    unsigned short* ff_bf  = bf;                bf += (long)M * NFF;
    unsigned short* pr_bf  = bf;                bf += (long)NB * NHEADS * NS * NS;
    float* sc = (float*)bf;                     // B*NH*S*S fp32 scores

    // ---- weights / dist_emb fp32 -> bf16, once per launch ----
    auto cvt = [&](const float* src, unsigned short* dst, long n) {
        cvt_bf16_kernel<<<(unsigned)((n / 8 + 255) / 256), 256, 0, stream>>>(src, dst, n);
    };
    cvt(qw, qw_bf, W1); cvt(kw, kw_bf, W1); cvt(vw, vw_bf, W1);
    cvt(attn_out_w, ow_bf, W1); cvt(int_w, iw_bf, W2); cvt(out_w, pw_bf, W2);
    cvt(dist_emb, dist_bf, WD);

    // embeddings -> h
    embed_kernel<<<M, 256, 0, stream>>>(input_ids, tok_type, word_emb, tok_emb, h);

    const long SH  = (long)NS * NHID;      // per-batch stride, [B,S,H] views
    const long SHD = (long)NS * NHD;       // per-(b,h) stride, [B,NH,S,HD] views
    const long SS  = (long)NS * NS;
    for (int l = 0; l < NL; ++l) {
        const unsigned short* qw_l = qw_bf + (long)l * NHID * NHID;
        const unsigned short* kw_l = kw_bf + (long)l * NHID * NHID;
        const unsigned short* vw_l = vw_bf + (long)l * NHID * NHID;
        const unsigned short* ow_l = ow_bf + (long)l * NHID * NHID;
        const unsigned short* iw_l = iw_bf + (long)l * NFF * NHID;
        const unsigned short* pw_l = pw_bf + (long)l * NHID * NFF;
        const unsigned short* dist_l = dist_bf + (long)l * (2 * NMAXPOS - 1) * NHD;

        // x = LN(h; attn_ln)  (bf16 out)
        layernorm_kernel<<<M, 256, 0, stream>>>(h, attn_ln_w + l * NHID,
                                                attn_ln_b + l * NHID, x_bf);
        dim3 gProj(NHID / 64, M / 128, 1);
        // q/k = x @ W^T + b -> [B,NH,S,HD] bf16 (CBHSD epilogue, TDM staging)
        gemm_wmma<128, 64, true, false, false, true, true, true><<<gProj, 256, 0, stream>>>(
            x_bf, qw_l, q_bf, qb + l * NHID, nullptr, NHID, NHID, NHID, NHID, 1,
            1, 0, 0, 1, 0, 0, 1, 0, 0);
        gemm_wmma<128, 64, true, false, false, true, true, true><<<gProj, 256, 0, stream>>>(
            x_bf, kw_l, k_bf, kb + l * NHID, nullptr, NHID, NHID, NHID, NHID, 1,
            1, 0, 0, 1, 0, 0, 1, 0, 0);
        // v = x @ W^T + b -> [B,S,H] bf16 (TDM staging)
        gemm_wmma<128, 64, true, false, false, true, false, true><<<gProj, 256, 0, stream>>>(
            x_bf, vw_l, v_bf, vb + l * NHID, nullptr, NHID, NHID, NHID, NHID, 1,
            1, 0, 0, 1, 0, 0, 1, 0, 0);

        // scores[z] = q_z @ k_z^T, z = b*NH+h (K=64), fp32 out
        dim3 gScore(NS / 64, NS / 128, NB * NHEADS);
        gemm_wmma<128, 64, true, false, false, false, false, false><<<gScore, 256, 0, stream>>>(
            q_bf, k_bf, sc, nullptr, nullptr, NHD, NHD, NHD, NS, 1,
            1, SHD, 0, 1, SHD, 0, 1, SS, 0);

        // scores += q . pe   (per-l batched GEMM, B = dist view, ldb = -64)
        dim3 gRel(NS / 128, (NB * NHEADS) / 64, NS);
        gemm_wmma<64, 128, true, false, true, false, false, false><<<gRel, 256, 0, stream>>>(
            q_bf, dist_l + (long)(NMAXPOS - 1) * NHD, sc, nullptr, sc,
            NHD, (int)SHD, -NHD, (int)SS, 1,
            1, NHD, 0,               // A: + l*HD
            1, NHD, 0,               // B: + l*64  (row r at (l-r+511)*64)
            1, NS, 0);               // C: cOff = l*S ; off = m*S*S + l*S + r
        // scores += k . pe   (per-r batched GEMM, B = dist view, ldb = +64)
        gemm_wmma<64, 128, true, false, true, false, false, false><<<gRel, 256, 0, stream>>>(
            k_bf, dist_l + (long)(NMAXPOS - 1) * NHD, sc, nullptr, sc,
            NHD, (int)SHD, NHD, (int)SS, NS,
            1, NHD, 0,               // A: + r*HD
            1, -NHD, 0,              // B: - r*64  (row l at (l-r+511)*64)
            1, 1, 0);                // C: cOff = r ; off = m*S*S + l*S + r

        // probs = softmax(scale * scores + mask)  (bf16 out)
        softmax_kernel<<<(NB * NHEADS * NS) / 8, 256, 0, stream>>>(sc, attn_mask, pr_bf);

        // ctx[z] = probs_z @ v_z  -> [B,S,H] bf16 via head-view C mapping
        dim3 gCtx(NHD / 64, NS / 128, NB * NHEADS);
        gemm_wmma<128, 64, false, false, false, true, false, false><<<gCtx, 256, 0, stream>>>(
            pr_bf, v_bf, ctx_bf, nullptr, nullptr, NS, NS, NHID, NHID, 1,
            1, SS, 0,                // A: probs
            NHEADS, SH, NHD,         // B: v head view ([K,N] row-major)
            NHEADS, SH, NHD);        // C: ctx head view

        // h = h + ctx @ attn_out_w^T + attn_out_b   (fp32 out, residual, TDM)
        gemm_wmma<128, 64, true, false, true, false, false, true><<<gProj, 256, 0, stream>>>(
            ctx_bf, ow_l, h, attn_out_b + l * NHID, h, NHID, NHID, NHID, NHID, 1,
            1, 0, 0, 1, 0, 0, 1, 0, 0);

        // x = LN(h; ffn_ln)  (bf16 out)
        layernorm_kernel<<<M, 256, 0, stream>>>(h, ffn_ln_w + l * NHID,
                                                ffn_ln_b + l * NHID, x_bf);
        // ff = gelu(x @ int_w^T + int_b)  (bf16 out, TDM)
        dim3 gFF1(NFF / 64, M / 128, 1);
        gemm_wmma<128, 64, true, true, false, true, false, true><<<gFF1, 256, 0, stream>>>(
            x_bf, iw_l, ff_bf, int_b + l * NFF, nullptr, NHID, NHID, NHID, NFF, 1,
            1, 0, 0, 1, 0, 0, 1, 0, 0);
        // h = h + ff @ out_w^T + out_b   (fp32 out, residual, TDM)
        dim3 gFF2(NHID / 64, M / 128, 1);
        gemm_wmma<128, 64, true, false, true, false, false, true><<<gFF2, 256, 0, stream>>>(
            ff_bf, pw_l, h, out_b + l * NHID, h, NFF, NFF, NFF, NHID, 1,
            1, 0, 0, 1, 0, 0, 1, 0, 0);
    }
    (void)in_sizes; (void)n_in; (void)out_size; (void)ws_size;
}
